// NF4Linear_62277025792623
// MI455X (gfx1250) — matrix operand order repaired and checked by reading, not compiled, not run on hardware
//
#include <hip/hip_runtime.h>

// ---------------------------------------------------------------------------
// NF4-quantized linear: y = x @ W^T + bias      (M=4096, K=4096, N=14336)
// Compute-bound (481 GFLOP vs ~535 MB min traffic) -> bf16 WMMA path.
// Round 2: software-pipelined, double-buffered LDS so the per-tile
// global_load_b128 latency hides under the 16 WMMAs of the previous tile.
// ---------------------------------------------------------------------------

typedef __attribute__((ext_vector_type(16))) __bf16 v16bf;
typedef __attribute__((ext_vector_type(8)))  float  v8f;

#define IN_F   4096
#define OUT_F  14336
#define M_DIM  4096

#define BM     128
#define BN     128
#define BK     64            // one NF4 quant block; two WMMA k-steps of 32
#define LDSW   72            // LDS row stride in bf16 elements (64 + 8 pad)
#define BUFSZ  ((BM + BN) * LDSW)   // one ping-pong buffer (x tile + w tile)
#define NKIT   (IN_F / BK)

__device__ __constant__ float NF4_TAB[16] = {
    -1.0f, -0.6961928009986877f, -0.5250730514526367f, -0.39491748809814453f,
    -0.28444138169288635f, -0.18477343022823334f, -0.09105003625154495f, 0.0f,
    0.07958029955625534f, 0.16093020141124725f, 0.24611230194568634f,
    0.33791524171829224f, 0.44070982933044434f, 0.5626170039176941f,
    0.8333911895751953f, 1.0f};

static __device__ __forceinline__ unsigned pk_bf16(float lo, float hi) {
  union { __bf16 h[2]; unsigned u; } r;
  r.h[0] = (__bf16)lo;
  r.h[1] = (__bf16)hi;
  return r.u;
}

// Codebook lookup via cross-lane permute: lanes hold NF4_TAB[lane&15];
// idx in 0..15 selects lane idx (ds_bpermute byte index = idx*4).
static __device__ __forceinline__ float nf4_lut(int idx, int codeBits) {
  return __int_as_float(__builtin_amdgcn_ds_bpermute(idx << 2, codeBits));
}

// Per-thread staging registers for one K-tile (issued early, consumed late).
struct Staging {
  float4 xr[8];   // 32 fp32 x elements
  int4   wr[8];   // 32 int32 NF4 indices
  float  s;       // per-row absmax for this 64-wide K block
};

static __device__ __forceinline__ void issue_tile_loads(
    const float* __restrict__ xbase, const int* __restrict__ wbase,
    const float* __restrict__ abase, int k0, Staging& st) {
  const float4* xsrc = (const float4*)(xbase + k0);
  const int4*   wsrc = (const int4*)(wbase + k0);
#pragma unroll
  for (int i = 0; i < 8; ++i) st.xr[i] = xsrc[i];
#pragma unroll
  for (int i = 0; i < 8; ++i) st.wr[i] = wsrc[i];
  st.s = abase[k0 >> 6];
}

static __device__ __forceinline__ void stage_to_lds(
    const Staging& st, __bf16* xs, __bf16* ws,
    int lrow, int lkh, int codeBits) {
  // x tile: fp32 -> bf16, 4x ds_store_b128
  uint4* xdst = (uint4*)&xs[lrow * LDSW + lkh];
#pragma unroll
  for (int i = 0; i < 4; ++i) {
    float4 a = st.xr[2 * i];
    float4 b = st.xr[2 * i + 1];
    uint4 p;
    p.x = pk_bf16(a.x, a.y);
    p.y = pk_bf16(a.z, a.w);
    p.z = pk_bf16(b.x, b.y);
    p.w = pk_bf16(b.z, b.w);
    xdst[i] = p;
  }
  // w tile: NF4 dequant (bpermute LUT * absmax) -> bf16, 4x ds_store_b128
  uint4* wdst = (uint4*)&ws[lrow * LDSW + lkh];
#pragma unroll
  for (int i = 0; i < 4; ++i) {
    int4 q0 = st.wr[2 * i];
    int4 q1 = st.wr[2 * i + 1];
    float f0 = nf4_lut(q0.x, codeBits) * st.s;
    float f1 = nf4_lut(q0.y, codeBits) * st.s;
    float f2 = nf4_lut(q0.z, codeBits) * st.s;
    float f3 = nf4_lut(q0.w, codeBits) * st.s;
    float f4 = nf4_lut(q1.x, codeBits) * st.s;
    float f5 = nf4_lut(q1.y, codeBits) * st.s;
    float f6 = nf4_lut(q1.z, codeBits) * st.s;
    float f7 = nf4_lut(q1.w, codeBits) * st.s;
    uint4 p;
    p.x = pk_bf16(f0, f1);
    p.y = pk_bf16(f2, f3);
    p.z = pk_bf16(f4, f5);
    p.w = pk_bf16(f6, f7);
    wdst[i] = p;
  }
}

static __device__ __forceinline__ void compute_tile(
    const __bf16* xs, const __bf16* ws, v8f (*acc)[2],
    int arow0, int akb0, int bcol0, int bkb0) {
#pragma unroll
  for (int kk = 0; kk < BK; kk += 32) {
    v16bf afrag[4];
#pragma unroll
    for (int mt = 0; mt < 4; ++mt) {
      const __bf16* p = &xs[(arow0 + mt * 16) * LDSW + akb0 + kk];
      union { uint4 u[2]; v16bf v; } t;
      t.u[0] = *(const uint4*)p;          // K = kb .. kb+7
      t.u[1] = *(const uint4*)(p + 16);   // K = kb+16 .. kb+23
      afrag[mt] = t.v;
    }
    v16bf bfrag[2];
#pragma unroll
    for (int nt = 0; nt < 2; ++nt) {
      const __bf16* p = &ws[(bcol0 + nt * 16) * LDSW + bkb0 + kk];
      union { uint4 u[2]; v16bf v; } t;
      t.u[0] = *(const uint4*)p;          // K = kb .. kb+7
      t.u[1] = *(const uint4*)(p + 16);   // K = kb+16 .. kb+23
      bfrag[nt] = t.v;
    }
#pragma unroll
    for (int mt = 0; mt < 4; ++mt)
#pragma unroll
      for (int nt = 0; nt < 2; ++nt)
        acc[mt][nt] = __builtin_amdgcn_wmma_f32_16x16x32_bf16(
            false, afrag[mt], false, bfrag[nt],
            (short)0, acc[mt][nt], false, false);
  }
}

__global__ __launch_bounds__(256) void nf4_wmma_gemm(
    const float* __restrict__ x,
    const int*   __restrict__ w_idx,
    const float* __restrict__ absmax,
    const float* __restrict__ bias,
    float*       __restrict__ out)
{
  __shared__ __bf16 smem[2 * BUFSZ];    // ping-pong: [x-tile | w-tile] x2, 72 KB

  const int tid  = threadIdx.x;
  const int lane = tid & 31;
  const int wave = tid >> 5;     // 0..7
  const int wm   = wave >> 2;    // 0..1 : wave's M strip (64 rows)
  const int wn   = wave & 3;     // 0..3 : wave's N strip (32 cols)

  const int m0 = blockIdx.x * BM;   // fast-varying -> W band stays in L2
  const int n0 = blockIdx.y * BN;

  const int codeBits = __float_as_int(NF4_TAB[lane & 15]);

  // Cooperative loader mapping: thread t handles half of one tile row
  const int lrow = tid >> 1;           // 0..127
  const int lkh  = (tid & 1) << 5;     // 0 or 32

  const float* xbase = x      + (size_t)(m0 + lrow) * IN_F + lkh;
  const int*   wbase = w_idx  + (size_t)(n0 + lrow) * IN_F + lkh;
  const float* abase = absmax + (size_t)(n0 + lrow) * (IN_F / 64);

  // Fragment addressing (ISA 7.12.2 16-bit A/B layouts)
  const int arow0 = wm * 64 + (lane & 15);
  const int akb0  = (lane & 16) ? 8 : 0;
  const int bcol0 = wn * 32 + (lane & 15);
  const int bkb0  = (lane & 16) ? 8 : 0;

  v8f acc[4][2];
#pragma unroll
  for (int mt = 0; mt < 4; ++mt)
#pragma unroll
    for (int nt = 0; nt < 2; ++nt)
      acc[mt][nt] = (v8f){0.f, 0.f, 0.f, 0.f, 0.f, 0.f, 0.f, 0.f};

  // ---- prologue: fill buffer 0 ----
  Staging st;
  issue_tile_loads(xbase, wbase, abase, 0, st);
  stage_to_lds(st, smem, smem + BM * LDSW, lrow, lkh, codeBits);
  __syncthreads();

  // ---- pipelined main loop ----
  int p = 0;
  for (int it = 1; it < NKIT; ++it) {
    // issue next tile's global loads now; latency hides under compute below
    issue_tile_loads(xbase, wbase, abase, it * BK, st);
    // warm L2 two tiles ahead for the streaming quantized weights
    if (it + 1 < NKIT)
      __builtin_prefetch((const void*)(wbase + (it + 1) * BK), 0, 3);

    __bf16* cxs = smem + p * BUFSZ;
    compute_tile(cxs, cxs + BM * LDSW, acc, arow0, akb0, bcol0, bkb0);

    __bf16* sxs = smem + (p ^ 1) * BUFSZ;
    stage_to_lds(st, sxs, sxs + BM * LDSW, lrow, lkh, codeBits);
    __syncthreads();
    p ^= 1;
  }
  // ---- epilogue compute on last buffer ----
  {
    __bf16* cxs = smem + p * BUFSZ;
    compute_tile(cxs, cxs + BM * LDSW, acc, arow0, akb0, bcol0, bkb0);
  }

  // ---- writeback: C/D layout (VGPR r -> M=r / r+8; lane&15 -> N), + bias ----
  const int cn    = lane & 15;
  const int cmoff = (lane & 16) ? 8 : 0;
#pragma unroll
  for (int nt = 0; nt < 2; ++nt) {
    const int col  = n0 + wn * 32 + nt * 16 + cn;
    const float bv = bias[col];
#pragma unroll
    for (int mt = 0; mt < 4; ++mt) {
      const int rowb = m0 + wm * 64 + mt * 16 + cmoff;
      float* o = out + (size_t)rowb * OUT_F + col;
#pragma unroll
      for (int r = 0; r < 8; ++r)
        o[(size_t)r * OUT_F] = acc[mt][nt][r] + bv;
    }
  }
}

extern "C" void kernel_launch(void* const* d_in, const int* in_sizes, int n_in,
                              void* d_out, int out_size, void* d_ws, size_t ws_size,
                              hipStream_t stream) {
  (void)in_sizes; (void)n_in; (void)out_size; (void)d_ws; (void)ws_size;
  const float* x      = (const float*)d_in[0];
  const int*   w_idx  = (const int*)d_in[1];
  const float* absmax = (const float*)d_in[2];
  const float* bias   = (const float*)d_in[3];
  float* out = (float*)d_out;

  dim3 grid(M_DIM / BM, OUT_F / BN);   // (32, 112); x fast-varying -> W band L2-resident
  dim3 block(256);                     // 8 waves (wave32)
  hipLaunchKernelGGL(nf4_wmma_gemm, grid, block, 0, stream,
                     x, w_idx, absmax, bias, out);
}
